// SAE_18296560681188
// MI455X (gfx1250) — matrix operand order repaired and checked by reading, not compiled, not run on hardware
//
#include <hip/hip_runtime.h>
#include <hip/hip_bf16.h>

// Problem constants (match reference)
#define B_SZ   4096
#define D_SZ   1024
#define H_SZ   32768
#define KSEL   32

// Encoder tiling
#define MTILE     32          // rows per block (2 x 16 WMMA sub-tiles)
#define NCHUNK    256         // hidden columns per block iteration (8 waves * 2 * 16)
#define HSPLIT    2           // H halves handled by separate blocks, merged after
#define AS_STRIDE 1032        // bf16 elems per A row in LDS (1024 + 8 pad -> bank spread)
#define CT_STRIDE 260         // f32 elems per C row in LDS (256 + 4 pad)

typedef __attribute__((ext_vector_type(16))) __bf16 v16bf;
typedef __attribute__((ext_vector_type(8)))  __bf16 v8bf;
typedef __attribute__((ext_vector_type(4)))  __bf16 v4bf;
typedef __attribute__((ext_vector_type(8)))  float  v8f;
typedef __attribute__((ext_vector_type(4)))  int    v4i;

// ---------------- wave32 primitives ----------------
__device__ __forceinline__ unsigned wave_ballot32(bool p) {
#if __has_builtin(__builtin_amdgcn_ballot_w32)
  return __builtin_amdgcn_ballot_w32(p);
#else
  return (unsigned)__ballot(p);
#endif
}
__device__ __forceinline__ int bperm_i(int v, int srclane) {
  return __builtin_amdgcn_ds_bpermute(srclane << 2, v);
}
__device__ __forceinline__ float bperm_f(float v, int srclane) {
  return __int_as_float(__builtin_amdgcn_ds_bpermute(srclane << 2, __float_as_int(v)));
}
__device__ __forceinline__ float wave_min_f(float v, int lane) {
  #pragma unroll
  for (int off = 16; off > 0; off >>= 1) v = fminf(v, bperm_f(v, lane ^ off));
  return v;
}

// Stream one candidate per lane into the wave-resident top-32 list
// (one (val,idx) entry per lane; curmin is the wave-uniform list minimum).
__device__ __forceinline__ void topk_insert(float cand, int cidx,
                                            float& lval, int& lidx,
                                            float& curmin, int lane) {
  unsigned m = wave_ballot32(cand > curmin);
  while (m) {
    int src = __builtin_ctz(m);
    m &= m - 1;
    float cv = bperm_f(cand, src);
    int   ci = bperm_i(cidx, src);
    if (cv > curmin) {
      // wave argmin over list with total order (val, lane) -> unique winner
      float mv = lval; int ml = lane;
      #pragma unroll
      for (int off = 16; off > 0; off >>= 1) {
        float ov = bperm_f(mv, lane ^ off);
        int   ol = bperm_i(ml, lane ^ off);
        if (ov < mv || (ov == mv && ol < ml)) { mv = ov; ml = ol; }
      }
      if (lane == ml) { lval = cv; lidx = ci; }
      curmin = wave_min_f(lval, lane);
    }
  }
}

// ---------------- kernel 1: x fp32 -> bf16 ----------------
__global__ void __launch_bounds__(256)
k_convert_x(const float4* __restrict__ x4, __bf16* __restrict__ xb) {
  int i = blockIdx.x * 256 + threadIdx.x;     // B*D/4 elements of float4
  float4 v = x4[i];
  v4bf o;
  o[0] = (__bf16)v.x; o[1] = (__bf16)v.y; o[2] = (__bf16)v.z; o[3] = (__bf16)v.w;
  ((v4bf*)xb)[i] = o;
}

// ---------------- kernel 2: W_enc [D,H] fp32 -> WbT [H,D] bf16 ----------------
__global__ void __launch_bounds__(256)
k_transpose_wenc(const float* __restrict__ W, __bf16* __restrict__ WT) {
  __shared__ __bf16 tile[32][33];
  const int hb = blockIdx.x * 32;
  const int db = blockIdx.y * 32;
  const int tx = threadIdx.x;       // 0..31
  const int ty = threadIdx.y;       // 0..7
  #pragma unroll
  for (int i = ty; i < 32; i += 8)
    tile[i][tx] = (__bf16)W[(size_t)(db + i) * H_SZ + (hb + tx)];
  __syncthreads();
  #pragma unroll
  for (int i = ty; i < 32; i += 8)
    WT[(size_t)(hb + i) * D_SZ + (db + tx)] = tile[tx][i];
}

// ---------------- kernel 3: encoder GEMM + bias + ReLU + streaming top-32 ----
__global__ void __launch_bounds__(256)
k_encode_topk(const __bf16* __restrict__ xb, const __bf16* __restrict__ WbT,
              const float* __restrict__ b_enc,
              float* __restrict__ partv, int* __restrict__ parti) {
  __shared__ __attribute__((aligned(16))) __bf16 As[MTILE * AS_STRIDE];
  __shared__ __attribute__((aligned(16))) float  Ct[MTILE * CT_STRIDE];

  const int t     = threadIdx.x;
  const int wave  = t >> 5;
  const int lane  = t & 31;
  const int n16   = lane & 15;
  const int kh    = lane >> 4;          // K-half select (WMMA 16-bit layout)
  const int rbase = blockIdx.x * MTILE;
  const int part  = blockIdx.y;         // H half
  const int hbase = part * (H_SZ / HSPLIT);

  // Stage A tile (32 rows x 1024 bf16) into LDS via the async-to-LDS path
  // when available; plain vector copy otherwise.
  for (int i = t; i < MTILE * (D_SZ / 8); i += 256) {
    int r = i >> 7;                     // 128 16B-chunks per row
    int c = (i & 127) << 3;
    const __bf16* gsrc = &xb[(size_t)(rbase + r) * D_SZ + c];
    __bf16*       ldst = &As[r * AS_STRIDE + c];
#if __has_builtin(__builtin_amdgcn_global_load_async_to_lds_b128)
    __builtin_amdgcn_global_load_async_to_lds_b128(
        (__attribute__((address_space(1))) v4i*)(uintptr_t)gsrc,
        (__attribute__((address_space(3))) v4i*)(uintptr_t)ldst, 0, 0);
#else
    *(uint4*)ldst = *(const uint4*)gsrc;
#endif
  }
#if __has_builtin(__builtin_amdgcn_global_load_async_to_lds_b128)
#if __has_builtin(__builtin_amdgcn_s_wait_asynccnt)
  __builtin_amdgcn_s_wait_asynccnt(0);
#else
  asm volatile("s_wait_asynccnt 0x0" ::: "memory");
#endif
#endif
  __syncthreads();

  // Per-row top-32 list: one entry per lane, rows wave*4+q. post-ReLU >= 0,
  // so init val=0 makes unused slots contribute nothing in the decoder.
  float lval[4]; int lidx[4]; float curmin[4];
  #pragma unroll
  for (int q = 0; q < 4; ++q) { lval[q] = 0.f; lidx[q] = lane; curmin[q] = 0.f; }

  for (int hc = hbase; hc < hbase + H_SZ / HSPLIT; hc += NCHUNK) {
    const int ncol0  = hc + wave * 32 + n16;        // first N tile for this wave
    const float bias0 = b_enc[ncol0];
    const float bias1 = b_enc[ncol0 + 16];
    const __bf16* bptr0 = &WbT[(size_t)ncol0 * D_SZ + kh * 16];
    const __bf16* bptr1 = bptr0 + (size_t)16 * D_SZ;
    if (hc + NCHUNK < hbase + H_SZ / HSPLIT) {
      __builtin_prefetch(bptr0 + (size_t)NCHUNK * D_SZ, 0, 1);
      __builtin_prefetch(bptr1 + (size_t)NCHUNK * D_SZ, 0, 1);
    }

    v8f acc00 = {}, acc01 = {}, acc10 = {}, acc11 = {};
    const int kh8 = kh * 8;
    #pragma unroll 2
    for (int kb = 0; kb < D_SZ; kb += 32) {
      // B fragments: 16 contiguous bf16 of WbT row (K = kb + kh*16 + 0..15)
      v16bf bfrag0 = *(const v16bf*)(bptr0 + kb);
      v16bf bfrag1 = *(const v16bf*)(bptr1 + kb);
      // A fragments (contiguous K, matching B): elems 0..15 -> K=kb+kh*16..+15,
      // row m = lane&15 (sub-tiles 0/1).
      v8bf a0l = *(const v8bf*)&As[n16 * AS_STRIDE + kb + kh8 * 2];
      v8bf a0h = *(const v8bf*)&As[n16 * AS_STRIDE + kb + 8 + kh8 * 2];
      v8bf a1l = *(const v8bf*)&As[(n16 + 16) * AS_STRIDE + kb + kh8 * 2];
      v8bf a1h = *(const v8bf*)&As[(n16 + 16) * AS_STRIDE + kb + 8 + kh8 * 2];
      v16bf a0 = __builtin_shufflevector(a0l, a0h,
          0, 1, 2, 3, 4, 5, 6, 7, 8, 9, 10, 11, 12, 13, 14, 15);
      v16bf a1 = __builtin_shufflevector(a1l, a1h,
          0, 1, 2, 3, 4, 5, 6, 7, 8, 9, 10, 11, 12, 13, 14, 15);
      acc00 = __builtin_amdgcn_wmma_f32_16x16x32_bf16(false, a0, false, bfrag0,
                                                      (short)0, acc00, false, false);
      acc10 = __builtin_amdgcn_wmma_f32_16x16x32_bf16(false, a1, false, bfrag0,
                                                      (short)0, acc10, false, false);
      acc01 = __builtin_amdgcn_wmma_f32_16x16x32_bf16(false, a0, false, bfrag1,
                                                      (short)0, acc01, false, false);
      acc11 = __builtin_amdgcn_wmma_f32_16x16x32_bf16(false, a1, false, bfrag1,
                                                      (short)0, acc11, false, false);
    }

    // bias + ReLU, scatter C tiles to LDS (row = v + kh*8 [+16], col in chunk)
    #pragma unroll
    for (int v = 0; v < 8; ++v) {
      const int r0  = v + kh * 8;
      const int col = wave * 32 + n16;
      Ct[r0 * CT_STRIDE + col]             = fmaxf(acc00[v] + bias0, 0.f);
      Ct[(r0 + 16) * CT_STRIDE + col]      = fmaxf(acc10[v] + bias0, 0.f);
      Ct[r0 * CT_STRIDE + col + 16]        = fmaxf(acc01[v] + bias1, 0.f);
      Ct[(r0 + 16) * CT_STRIDE + col + 16] = fmaxf(acc11[v] + bias1, 0.f);
    }
    __syncthreads();

    // Streaming top-32 merge: wave handles rows wave*4 + {0..3}.
    #pragma unroll
    for (int q = 0; q < 4; ++q) {
      const int r = wave * 4 + q;
      #pragma unroll
      for (int j = 0; j < NCHUNK / 32; ++j) {
        const int cloc = lane + 32 * j;
        topk_insert(Ct[r * CT_STRIDE + cloc], hc + cloc,
                    lval[q], lidx[q], curmin[q], lane);
      }
    }
    __syncthreads();
  }

  #pragma unroll
  for (int q = 0; q < 4; ++q) {
    const int row = rbase + wave * 4 + q;
    partv[((size_t)part * B_SZ + row) * KSEL + lane] = lval[q];
    parti[((size_t)part * B_SZ + row) * KSEL + lane] = lidx[q];
  }
}

// ---------------- kernel 4: merge per-row partial top-32 lists ----------------
__global__ void __launch_bounds__(256)
k_merge_topk(const float* __restrict__ partv, const int* __restrict__ parti,
             float* __restrict__ topv, int* __restrict__ topi) {
  const int wave = threadIdx.x >> 5;
  const int lane = threadIdx.x & 31;
  const int row  = blockIdx.x * 8 + wave;
  // Part 0 is the initial list.
  float lval = partv[(size_t)row * KSEL + lane];
  int   lidx = parti[(size_t)row * KSEL + lane];
  float curmin = wave_min_f(lval, lane);
  // Stream part 1's 32 candidates.
  const float cand = partv[((size_t)B_SZ + row) * KSEL + lane];
  const int   cidx = parti[((size_t)B_SZ + row) * KSEL + lane];
  topk_insert(cand, cidx, lval, lidx, curmin, lane);
  topv[(size_t)row * KSEL + lane] = lval;
  topi[(size_t)row * KSEL + lane] = lidx;
}

// ---------------- kernel 5: sparse decode ----------------
__global__ void __launch_bounds__(256)
k_decode(const float* __restrict__ topv, const int* __restrict__ topi,
         const float* __restrict__ s, const float* __restrict__ W_dec,
         const float* __restrict__ b_dec, float* __restrict__ out) {
  __shared__ float sv[KSEL];
  __shared__ int   si[KSEL];
  const int row = blockIdx.x;
  const int t   = threadIdx.x;
  if (t < KSEL) {
    int   i = topi[row * KSEL + t];
    sv[t] = topv[row * KSEL + t] * s[i];
    si[t] = i;
  }
  __syncthreads();
  const int d = t * 4;                   // 256 threads * 4 = D
  float4 acc = *(const float4*)&b_dec[d];
  #pragma unroll 8
  for (int j = 0; j < KSEL; ++j) {
    const float  c = sv[j];
    const float4 w = *(const float4*)&W_dec[(size_t)si[j] * D_SZ + d];
    acc.x += c * w.x; acc.y += c * w.y; acc.z += c * w.z; acc.w += c * w.w;
  }
  *(float4*)&out[(size_t)row * D_SZ + d] = acc;
}

// ---------------- launch ----------------
extern "C" void kernel_launch(void* const* d_in, const int* in_sizes, int n_in,
                              void* d_out, int out_size, void* d_ws, size_t ws_size,
                              hipStream_t stream) {
  const float* x     = (const float*)d_in[0];
  const float* W_enc = (const float*)d_in[1];
  const float* b_enc = (const float*)d_in[2];
  const float* s     = (const float*)d_in[3];
  const float* W_dec = (const float*)d_in[4];
  const float* b_dec = (const float*)d_in[5];
  // d_in[6] == k (32) — baked into KSEL.

  const size_t XB_BYTES = (size_t)B_SZ * D_SZ * 2;                 // 8 MB
  const size_t WT_BYTES = (size_t)H_SZ * D_SZ * 2;                 // 64 MB
  const size_t PV_BYTES = (size_t)HSPLIT * B_SZ * KSEL * 4;        // 1 MB
  const size_t TV_BYTES = (size_t)B_SZ * KSEL * 4;                 // 0.5 MB

  char* ws = (char*)d_ws;
  __bf16* xb    = (__bf16*)(ws);
  __bf16* WbT   = (__bf16*)(ws + XB_BYTES);
  float*  partv = (float*)(ws + XB_BYTES + WT_BYTES);
  int*    parti = (int*)(ws + XB_BYTES + WT_BYTES + PV_BYTES);
  float*  topv  = (float*)(ws + XB_BYTES + WT_BYTES + 2 * PV_BYTES);
  int*    topi  = (int*)(ws + XB_BYTES + WT_BYTES + 2 * PV_BYTES + TV_BYTES);

  k_convert_x<<<(B_SZ * D_SZ / 4) / 256, 256, 0, stream>>>((const float4*)x, xb);
  k_transpose_wenc<<<dim3(H_SZ / 32, D_SZ / 32), dim3(32, 8), 0, stream>>>(W_enc, WbT);
  k_encode_topk<<<dim3(B_SZ / MTILE, HSPLIT), 256, 0, stream>>>(xb, WbT, b_enc,
                                                                partv, parti);
  k_merge_topk<<<B_SZ / 8, 256, 0, stream>>>(partv, parti, topv, topi);
  k_decode<<<B_SZ, 256, 0, stream>>>(topv, topi, s, W_dec, b_dec, (float*)d_out);
}
